// CausalSelfAttention_62835371540517
// MI455X (gfx1250) — matrix operand order repaired and checked by reading, not compiled, not run on hardware
//
#include <hip/hip_runtime.h>

// ---- types ----------------------------------------------------------------
typedef __attribute__((ext_vector_type(16))) __bf16 v16bf;
typedef __attribute__((ext_vector_type(8)))  float  v8f;
typedef __attribute__((ext_vector_type(4)))  unsigned int v4u;

union FragBf { v16bf v; v4u q[2]; };

#define WMMA_BF16(A, B, C) \
    __builtin_amdgcn_wmma_f32_16x16x32_bf16(false, (A), false, (B), (short)0, (C), false, false)

static constexpr int Bb = 4, Tt = 2048, Ee = 1024, Hh = 16, Dd = 64;

// f32 -> bf16 round-to-nearest-even
__device__ __forceinline__ unsigned short f2bf(float f) {
    unsigned u = __float_as_uint(f);
    u += 0x7FFFu + ((u >> 16) & 1u);
    return (unsigned short)(u >> 16);
}

// ---- kernel 1: f32 -> bf16 conversion -------------------------------------
__global__ void cvt_bf16_kernel(const float* __restrict__ src,
                                unsigned short* __restrict__ dst, int n) {
    int i = blockIdx.x * blockDim.x + threadIdx.x;
    int stride = gridDim.x * blockDim.x;
    for (; i < n; i += stride) dst[i] = f2bf(src[i]);
}

// ---- kernel 2: QKV projection, 64x64 tile per wave ------------------------
// qkv[t,f] = sum_e x[t,e] * w[f,e];  Q scaled by 1/sqrt(D); V stored transposed.
__global__ __launch_bounds__(128) void qkv_gemm_kernel(
    const unsigned short* __restrict__ xb,   // [B*T, E] bf16
    const unsigned short* __restrict__ wb,   // [3E, E] bf16
    unsigned short* __restrict__ qp,         // [B*H, T, D] bf16
    unsigned short* __restrict__ kp,         // [B*H, T, D] bf16
    unsigned short* __restrict__ vtp) {      // [B*H, D, T] bf16
    const int wave = blockIdx.x * 4 + (threadIdx.x >> 5);
    const int lane = threadIdx.x & 31;
    const int ln = lane & 15, lh = lane >> 4;
    const int mt = wave / 48;                // 128 macro row tiles (64 rows)
    const int nt = wave % 48;                // 48 macro col tiles (64 cols)
    const int mBase = mt * 64, nBase = nt * 64;

    const unsigned short* arow[4];
    const unsigned short* brow[4];
#pragma unroll
    for (int s = 0; s < 4; ++s) {
        arow[s] = xb + (size_t)(mBase + s * 16 + ln) * Ee;
        brow[s] = wb + (size_t)(nBase + s * 16 + ln) * Ee;
    }

    v8f acc[4][4] = {};
    for (int ke = 0; ke < Ee; ke += 32) {
        FragBf a[4], b[4];
#pragma unroll
        for (int s = 0; s < 4; ++s) {
            a[s].q[0] = *(const v4u*)(arow[s] + ke + lh * 8);
            a[s].q[1] = *(const v4u*)(arow[s] + ke + 16 + lh * 8);
            b[s].q[0] = *(const v4u*)(brow[s] + ke + lh * 8);
            b[s].q[1] = *(const v4u*)(brow[s] + ke + 16 + lh * 8);
        }
#pragma unroll
        for (int ms = 0; ms < 4; ++ms)
#pragma unroll
            for (int ns = 0; ns < 4; ++ns)
                acc[ms][ns] = WMMA_BF16(a[ms].v, b[ns].v, acc[ms][ns]);
    }

#pragma unroll
    for (int ns = 0; ns < 4; ++ns) {
        const int fcol = nBase + ns * 16 + ln;
        const int which = fcol >> 10;        // 0=Q 1=K 2=V
        const int fr = fcol & 1023;
        const int h = fr >> 6, d = fr & 63;
#pragma unroll
        for (int ms = 0; ms < 4; ++ms) {
#pragma unroll
            for (int r = 0; r < 8; ++r) {
                const int rg = mBase + ms * 16 + lh * 8 + r;  // flattened B*T
                const int b_ = rg >> 11, t_ = rg & 2047;
                const int bh = b_ * Hh + h;
                const float v = acc[ms][ns][r];
                if (which == 0)      qp[((size_t)bh * Tt + t_) * Dd + d] = f2bf(v * 0.125f);
                else if (which == 1) kp[((size_t)bh * Tt + t_) * Dd + d] = f2bf(v);
                else                 vtp[((size_t)bh * Dd + d) * Tt + t_] = f2bf(v);
            }
        }
    }
}

// ---- kernel 3: flash attention (one wave per 32-row query tile) -----------
__global__ __launch_bounds__(128) void attn_kernel(
    const unsigned short* __restrict__ qp,   // [B*H, T, D] bf16 (pre-scaled)
    const unsigned short* __restrict__ kp,   // [B*H, T, D] bf16
    const unsigned short* __restrict__ vtp,  // [B*H, D, T] bf16
    unsigned short* __restrict__ yb) {       // [B*T, E] bf16
    __shared__ __align__(16) unsigned short lds[4][32 * 32];

    const int wslot = threadIdx.x >> 5;
    const int wave = blockIdx.x * 4 + wslot;
    const int lane = threadIdx.x & 31;
    const int ln = lane & 15, lh = lane >> 4;
    const int qt = wave & 63;        // 64 query tiles of 32 rows
    const int bh = wave >> 6;        // 64 (b,h) pairs
    const int qBase = qt * 32;

    // Q A-fragments: [query subtile][d half], resident for the whole loop
    FragBf aq[2][2];
#pragma unroll
    for (int qi = 0; qi < 2; ++qi) {
        const unsigned short* qrow = qp + ((size_t)bh * Tt + qBase + qi * 16 + ln) * Dd;
        aq[qi][0].q[0] = *(const v4u*)(qrow + lh * 8);
        aq[qi][0].q[1] = *(const v4u*)(qrow + 16 + lh * 8);
        aq[qi][1].q[0] = *(const v4u*)(qrow + 32 + lh * 8);
        aq[qi][1].q[1] = *(const v4u*)(qrow + 48 + lh * 8);
    }

    v8f o[2][4] = {};
    float mrun[2][8], lrun[2][8];
#pragma unroll
    for (int qi = 0; qi < 2; ++qi)
#pragma unroll
        for (int r = 0; r < 8; ++r) { mrun[qi][r] = -1e30f; lrun[qi][r] = 0.0f; }

    const unsigned short* kbp = kp + (size_t)bh * Tt * Dd;
    const unsigned short* vbp = vtp + (size_t)bh * Dd * Tt;
    unsigned short* myLds = lds[wslot];

    for (int kb = 0; kb < qBase + 32; kb += 32) {
        // ---- K fragments: [kv subtile][d half] ----
        FragBf bk[2][2];
#pragma unroll
        for (int t = 0; t < 2; ++t) {
            const unsigned short* kr = kbp + (size_t)(kb + t * 16 + ln) * Dd;
            bk[t][0].q[0] = *(const v4u*)(kr + lh * 8);
            bk[t][0].q[1] = *(const v4u*)(kr + 16 + lh * 8);
            bk[t][1].q[0] = *(const v4u*)(kr + 32 + lh * 8);
            bk[t][1].q[1] = *(const v4u*)(kr + 48 + lh * 8);
        }

        // ---- S = Q * K^T : 2 query subtiles x 2 kv subtiles ----
        v8f s[2][2] = {};
#pragma unroll
        for (int qi = 0; qi < 2; ++qi)
#pragma unroll
            for (int t = 0; t < 2; ++t) {
                s[qi][t] = WMMA_BF16(aq[qi][0].v, bk[t][0].v, s[qi][t]);
                s[qi][t] = WMMA_BF16(aq[qi][1].v, bk[t][1].v, s[qi][t]);
            }

        // ---- causal mask (diagonal block only; uniform branch) ----
        if (kb + 32 > qBase) {
#pragma unroll
            for (int qi = 0; qi < 2; ++qi)
#pragma unroll
                for (int t = 0; t < 2; ++t)
#pragma unroll
                    for (int r = 0; r < 8; ++r) {
                        const int row = qBase + qi * 16 + lh * 8 + r;
                        if (kb + t * 16 + ln > row) s[qi][t][r] = -1e30f;
                    }
        }

        // ---- online softmax per query subtile ----
#pragma unroll
        for (int qi = 0; qi < 2; ++qi) {
            float rm[8];
#pragma unroll
            for (int r = 0; r < 8; ++r) rm[r] = fmaxf(s[qi][0][r], s[qi][1][r]);
#pragma unroll
            for (int off = 1; off < 16; off <<= 1)
#pragma unroll
                for (int r = 0; r < 8; ++r)
                    rm[r] = fmaxf(rm[r], __shfl_xor(rm[r], off, 32));

            float rs[8];
#pragma unroll
            for (int r = 0; r < 8; ++r) {
                const float nm = fmaxf(mrun[qi][r], rm[r]);
                const float fac = __expf(mrun[qi][r] - nm);
                mrun[qi][r] = nm;
                s[qi][0][r] = __expf(s[qi][0][r] - nm);
                s[qi][1][r] = __expf(s[qi][1][r] - nm);
                rs[r] = s[qi][0][r] + s[qi][1][r];
                lrun[qi][r] *= fac;
                o[qi][0][r] *= fac; o[qi][1][r] *= fac;
                o[qi][2][r] *= fac; o[qi][3][r] *= fac;
            }
#pragma unroll
            for (int off = 1; off < 16; off <<= 1)
#pragma unroll
                for (int r = 0; r < 8; ++r)
                    rs[r] += __shfl_xor(rs[r], off, 32);
#pragma unroll
            for (int r = 0; r < 8; ++r) lrun[qi][r] += rs[r];

            // ---- transpose P (D-layout -> A-layout) through per-wave LDS --
#pragma unroll
            for (int r = 0; r < 8; ++r) {
                const int row = qi * 16 + lh * 8 + r;
                myLds[row * 32 + ln]      = f2bf(s[qi][0][r]);
                myLds[row * 32 + 16 + ln] = f2bf(s[qi][1][r]);
            }
        }
        asm volatile("s_wait_dscnt 0x0" ::: "memory"); // cross-lane LDS RAW

        FragBf ap[2];
#pragma unroll
        for (int qi = 0; qi < 2; ++qi) {
            ap[qi].q[0] = *(const v4u*)(myLds + (qi * 16 + ln) * 32 + lh * 8);
            ap[qi].q[1] = *(const v4u*)(myLds + (qi * 16 + ln) * 32 + 16 + lh * 8);
        }

        // ---- O += P * V : each V fragment reused across both query tiles --
#pragma unroll
        for (int dt = 0; dt < 4; ++dt) {
            FragBf bv;
            const unsigned short* vq = vbp + (size_t)(dt * 16 + ln) * Tt + kb + lh * 8;
            bv.q[0] = *(const v4u*)(vq);
            bv.q[1] = *(const v4u*)(vq + 16);
            o[0][dt] = WMMA_BF16(ap[0].v, bv.v, o[0][dt]);
            o[1][dt] = WMMA_BF16(ap[1].v, bv.v, o[1][dt]);
        }
    }

    // ---- normalize and write y as bf16 [B*T, E] ----
    const int b_ = bh >> 4, h = bh & 15;
#pragma unroll
    for (int qi = 0; qi < 2; ++qi)
#pragma unroll
        for (int r = 0; r < 8; ++r) {
            const float inv = 1.0f / lrun[qi][r];
            const int t_ = qBase + qi * 16 + lh * 8 + r;
            unsigned short* yr = yb + ((size_t)(b_ * Tt + t_)) * Ee + h * Dd;
            yr[ln]      = f2bf(o[qi][0][r] * inv);
            yr[16 + ln] = f2bf(o[qi][1][r] * inv);
            yr[32 + ln] = f2bf(o[qi][2][r] * inv);
            yr[48 + ln] = f2bf(o[qi][3][r] * inv);
        }
}

// ---- kernel 4: output projection, 64x64 tile per wave, f32 result ---------
__global__ __launch_bounds__(128) void out_gemm_kernel(
    const unsigned short* __restrict__ ybp,  // [B*T, E] bf16
    const unsigned short* __restrict__ wbp,  // [E, E] bf16
    float* __restrict__ outp) {              // [B*T, E] f32
    const int wave = blockIdx.x * 4 + (threadIdx.x >> 5);
    const int lane = threadIdx.x & 31;
    const int ln = lane & 15, lh = lane >> 4;
    const int mt = wave >> 4;                // 128 macro row tiles
    const int nt = wave & 15;                // 16 macro col tiles
    const int mBase = mt * 64, nBase = nt * 64;

    const unsigned short* arow[4];
    const unsigned short* brow[4];
#pragma unroll
    for (int s = 0; s < 4; ++s) {
        arow[s] = ybp + (size_t)(mBase + s * 16 + ln) * Ee;
        brow[s] = wbp + (size_t)(nBase + s * 16 + ln) * Ee;
    }

    v8f acc[4][4] = {};
    for (int ke = 0; ke < Ee; ke += 32) {
        FragBf a[4], b[4];
#pragma unroll
        for (int s = 0; s < 4; ++s) {
            a[s].q[0] = *(const v4u*)(arow[s] + ke + lh * 8);
            a[s].q[1] = *(const v4u*)(arow[s] + ke + 16 + lh * 8);
            b[s].q[0] = *(const v4u*)(brow[s] + ke + lh * 8);
            b[s].q[1] = *(const v4u*)(brow[s] + ke + 16 + lh * 8);
        }
#pragma unroll
        for (int ms = 0; ms < 4; ++ms)
#pragma unroll
            for (int ns = 0; ns < 4; ++ns)
                acc[ms][ns] = WMMA_BF16(a[ms].v, b[ns].v, acc[ms][ns]);
    }

#pragma unroll
    for (int ms = 0; ms < 4; ++ms)
#pragma unroll
        for (int ns = 0; ns < 4; ++ns)
#pragma unroll
            for (int r = 0; r < 8; ++r) {
                const int rg = mBase + ms * 16 + lh * 8 + r;
                outp[(size_t)rg * Ee + nBase + ns * 16 + ln] = acc[ms][ns][r];
            }
}

// ---- host side ------------------------------------------------------------
extern "C" void kernel_launch(void* const* d_in, const int* in_sizes, int n_in,
                              void* d_out, int out_size, void* d_ws, size_t ws_size,
                              hipStream_t stream) {
    const float* x     = (const float*)d_in[0];   // [4,2048,1024]
    const float* w_qkv = (const float*)d_in[1];   // [3072,1024]
    const float* w_out = (const float*)d_in[2];   // [1024,1024]
    float* out = (float*)d_out;

    char* ws = (char*)d_ws;
    const size_t nX = (size_t)Bb * Tt * Ee;       // 8,388,608
    const size_t nWqkv = (size_t)3 * Ee * Ee;     // 3,145,728
    const size_t nWout = (size_t)Ee * Ee;         // 1,048,576
    unsigned short* xb    = (unsigned short*)(ws);
    unsigned short* wqkvb = (unsigned short*)(ws + 16777216);
    unsigned short* woutb = (unsigned short*)(ws + 23068672);
    unsigned short* qp    = (unsigned short*)(ws + 25165824);
    unsigned short* kp    = (unsigned short*)(ws + 41943040);
    unsigned short* vtp   = (unsigned short*)(ws + 58720256);
    unsigned short* yb    = (unsigned short*)(ws + 75497472);

    cvt_bf16_kernel<<<2048, 256, 0, stream>>>(x, xb, (int)nX);
    cvt_bf16_kernel<<<2048, 256, 0, stream>>>(w_qkv, wqkvb, (int)nWqkv);
    cvt_bf16_kernel<<<1024, 256, 0, stream>>>(w_out, woutb, (int)nWout);

    // 128 x 48 macro tiles = 6144 waves / 4 per block
    qkv_gemm_kernel<<<1536, 128, 0, stream>>>(xb, wqkvb, qp, kp, vtp);

    // 64 (b,h) x 64 query tiles = 4096 waves / 4 per block
    attn_kernel<<<1024, 128, 0, stream>>>(qp, kp, vtp, yb);

    // 128 x 16 macro tiles = 2048 waves / 4 per block
    out_gemm_kernel<<<512, 128, 0, stream>>>(yb, woutb, out);
}
